// GroupedQueryAttention_45561013076092
// MI455X (gfx1250) — matrix operand order repaired and checked by reading, not compiled, not run on hardware
//
#include <hip/hip_runtime.h>
#include <hip/hip_bf16.h>
#include <math.h>

typedef __bf16 bf16_t;
typedef __attribute__((ext_vector_type(16))) __bf16 v16bf;
typedef __attribute__((ext_vector_type(8)))  __bf16 v8bf;
typedef __attribute__((ext_vector_type(8)))  float  v8f;

#define D_MODEL   1024
#define NUM_HEADS 16
#define NUM_GROUPS 4
#define DK        64
#define T_LEN     2048
#define BATCH     2
#define ROWS      (BATCH * T_LEN)   // 4096
#define WINDOW    512

// ---------- fragment loaders (contiguous k order chosen by this file, not the ISA table order; wave32) ----------
// B fragment (K x 16, 16-bit): lane l holds column n=l%16, K = (l/16)*16 + e,
// 16 contiguous bf16 (32 bytes).
static __device__ __forceinline__ v16bf load_b_frag(const bf16_t* p) {
    return *(const v16bf*)p;
}
// A fragment (16 x 32, 16-bit): lane l holds row m=l%16, K = (l/16)*16 + {0..15}:
// two adjacent 16-byte chunks.
static __device__ __forceinline__ v16bf load_a_frag(const bf16_t* p) {
    v8bf lo = *(const v8bf*)p;
    v8bf hi = *(const v8bf*)(p + 8);
    v16bf r;
#pragma unroll
    for (int i = 0; i < 8; ++i) { r[i] = lo[i]; r[i + 8] = hi[i]; }
    return r;
}

static __device__ __forceinline__ v8f wmma_bf16(v16bf a, v16bf b, v8f c) {
    // v_wmma_f32_16x16x32_bf16 : D = A(16x32) * B(32x16) + C
    return __builtin_amdgcn_wmma_f32_16x16x32_bf16(
        /*neg_a=*/false, a, /*neg_b=*/false, b,
        /*c_mod=*/(short)0, c, /*reuse_a=*/false, /*reuse_b=*/false);
}

static __device__ __forceinline__ v8f zero_v8f() {
    v8f z = {0.f, 0.f, 0.f, 0.f, 0.f, 0.f, 0.f, 0.f};
    return z;
}

// ---------- elementwise converts ----------
__global__ void cvt_f32_to_bf16(const float* __restrict__ in, bf16_t* __restrict__ out, int n) {
    int i = blockIdx.x * blockDim.x + threadIdx.x;
    if (i < n) out[i] = (bf16_t)in[i];
}

// RoPE (interleaved pairs) on fp32 input, bf16 output. feat = 1024 (Q) or 256 (K).
__global__ void rope_to_bf16(const float* __restrict__ in, bf16_t* __restrict__ out,
                             const int* __restrict__ pos, int feat, int total_pairs) {
    int idx = blockIdx.x * blockDim.x + threadIdx.x;
    if (idx >= total_pairs) return;
    int ppr = feat >> 1;              // pairs per row
    int r   = idx / ppr;
    int p2  = idx % ppr;              // pair index in row; f_even = 2*p2
    int dp  = p2 & 31;                // pair index within a 64-wide head
    int t   = r & (T_LEN - 1);
    float position = (float)pos[t];
    float inv_freq = __powf(10000.0f, -(float)dp * (1.0f / 32.0f));
    float s, c;
    __sincosf(position * inv_freq, &s, &c);
    size_t base = (size_t)r * feat + 2 * p2;
    float x1 = in[base], x2 = in[base + 1];
    out[base]     = (bf16_t)(x1 * c - x2 * s);
    out[base + 1] = (bf16_t)(x1 * s + x2 * c);
}

// V fp32 [b*t][256] -> Vt bf16 [b][g][64][t]  (so V^T loads contiguously along keys)
__global__ void v_transpose_bf16(const float* __restrict__ Vf, bf16_t* __restrict__ Vt, int n) {
    int i = blockIdx.x * blockDim.x + threadIdx.x;
    if (i >= n) return;
    int f  = i & 255;
    int r  = i >> 8;
    int t  = r & (T_LEN - 1);
    int bi = r >> 11;
    int g  = f >> 6;
    int d  = f & 63;
    Vt[(((size_t)bi * NUM_GROUPS + g) * DK + d) * T_LEN + t] = (bf16_t)Vf[i];
}

// ---------- WMMA GEMM: C[M,N] = A[M,K] * W[N,K]^T ----------
// One wave computes a 16x64 strip (A fragment reused across 4 N-tiles).
__global__ void gemm_bf16_f32(const bf16_t* __restrict__ A, const bf16_t* __restrict__ W,
                              float* __restrict__ C, int M, int N, int K) {
    const int lane = threadIdx.x & 31;
    const int wid  = threadIdx.x >> 5;
    const int l16  = lane & 15;
    const int lh   = lane >> 4;
    const int nstrips = N >> 6;
    int gw = blockIdx.x * (blockDim.x >> 5) + wid;
    if (gw >= (M >> 4) * nstrips) return;
    const int m0 = (gw / nstrips) << 4;
    const int n0 = (gw % nstrips) << 6;

    v8f c[4];
#pragma unroll
    for (int i = 0; i < 4; ++i) c[i] = zero_v8f();

    const bf16_t* arow = A + (size_t)(m0 + l16) * K + lh * 16;
    for (int k0 = 0; k0 < K; k0 += 32) {
        v16bf a = load_a_frag(arow + k0);
#pragma unroll
        for (int nn = 0; nn < 4; ++nn) {
            v16bf b = load_b_frag(W + (size_t)(n0 + nn * 16 + l16) * K + k0 + lh * 16);
            c[nn] = wmma_bf16(a, b, c[nn]);
        }
    }
    // D layout: lane holds col n = l%16, VGPR j -> row m = j + 8*(l/16)
#pragma unroll
    for (int nn = 0; nn < 4; ++nn) {
        int col = n0 + nn * 16 + l16;
#pragma unroll
        for (int j = 0; j < 8; ++j) {
            C[(size_t)(m0 + j + 8 * lh) * N + col] = c[nn][j];
        }
    }
}

// ---------- flash attention, sliding-window causal, 1 wave = 16 queries ----------
__global__ void attn_kernel(const bf16_t* __restrict__ Qb, const bf16_t* __restrict__ Kb,
                            const bf16_t* __restrict__ Vt, bf16_t* __restrict__ Ob) {
    const int lane = threadIdx.x & 31;
    const int wid  = threadIdx.x >> 5;
    const int l16  = lane & 15;
    const int lh   = lane >> 4;
    const int h  = blockIdx.y;        // head 0..15
    const int bi = blockIdx.z;        // batch
    const int g  = h >> 2;            // kv group
    const int qt = (blockIdx.x * 8 + wid) << 4;   // query tile base
    const int qi = qt + l16;          // this lane's query row (column of S^T)

    // Q B-fragments (loop invariant): B = Q^T (d x 16), d split into two K=32 steps
    const bf16_t* qrow = Qb + (size_t)(bi * T_LEN + qi) * D_MODEL + h * DK + lh * 16;
    const v16bf qb0 = load_b_frag(qrow);
    const v16bf qb1 = load_b_frag(qrow + 32);

    v8f o[4];
#pragma unroll
    for (int i = 0; i < 4; ++i) o[i] = zero_v8f();
    float m_run = -INFINITY, s_run = 0.f;

    int kb_start = qt - WINDOW;
    if (kb_start < 0) kb_start = 0;
    kb_start &= ~31;

    const bf16_t* Kbase = Kb + (size_t)(bi * T_LEN) * (NUM_GROUPS * DK) + g * DK;
    const bf16_t* Vbase = Vt + ((size_t)(bi * NUM_GROUPS + g) * DK) * T_LEN;

    for (int kb = kb_start; kb <= qt + 15; kb += 32) {
        // S^T = K * Q^T : two 16-key sub-tiles, K-dim = 64 in two steps of 32
        v8f s0 = zero_v8f(), s1 = zero_v8f();
        {
            const bf16_t* k0p = Kbase + (size_t)(kb + l16) * (NUM_GROUPS * DK) + lh * 16;
            const bf16_t* k1p = Kbase + (size_t)(kb + 16 + l16) * (NUM_GROUPS * DK) + lh * 16;
            v16bf a00 = load_a_frag(k0p);
            v16bf a01 = load_a_frag(k0p + 32);
            v16bf a10 = load_a_frag(k1p);
            v16bf a11 = load_a_frag(k1p + 32);
            s0 = wmma_bf16(a00, qb0, s0);
            s0 = wmma_bf16(a01, qb1, s0);
            s1 = wmma_bf16(a10, qb0, s1);
            s1 = wmma_bf16(a11, qb1, s1);
        }
        // scale + window mask; per lane: column q = qi, rows = key indices
        float f0[8], f1[8];
        float tmax = -INFINITY;
#pragma unroll
        for (int j = 0; j < 8; ++j) {
            int key0 = kb + j + 8 * lh;
            int key1 = key0 + 16;
            float v0 = s0[j] * 0.125f;
            float v1 = s1[j] * 0.125f;
            if (key0 > qi || qi - key0 > WINDOW) v0 = -INFINITY;
            if (key1 > qi || qi - key1 > WINDOW) v1 = -INFINITY;
            f0[j] = v0; f1[j] = v1;
            tmax = fmaxf(tmax, fmaxf(v0, v1));
        }
        tmax = fmaxf(tmax, __shfl_xor(tmax, 16, 32));
        float nm = fmaxf(m_run, tmax);
        bool dead = (nm == -INFINITY);
        float alpha = dead ? 0.f : __expf(m_run - nm);
        float p0[8], p1[8];
        float tsum = 0.f;
#pragma unroll
        for (int j = 0; j < 8; ++j) {
            float e0 = dead ? 0.f : __expf(f0[j] - nm);
            float e1 = dead ? 0.f : __expf(f1[j] - nm);
            p0[j] = e0; p1[j] = e1;
            tsum += e0 + e1;
        }
        tsum += __shfl_xor(tsum, 16, 32);
        s_run = s_run * alpha + tsum;
        m_run = nm;
#pragma unroll
        for (int dt = 0; dt < 4; ++dt)
#pragma unroll
            for (int j = 0; j < 8; ++j) o[dt][j] *= alpha;

        // Assemble P^T B-fragment (32 keys x 16 queries): lane needs 16 keys for
        // its query; swap 8 values with partner lane (l ^ 16).
        v16bf pb;
#pragma unroll
        for (int j = 0; j < 8; ++j) {
            float send = (lh == 0) ? p1[j] : p0[j];
            float recv = __shfl_xor(send, 16, 32);
            pb[j]     = (bf16_t)((lh == 0) ? p0[j] : recv);
            pb[j + 8] = (bf16_t)((lh == 0) ? recv : p1[j]);
        }
        // O^T += V^T * P^T : 4 d-row tiles of 16, K=32 keys
        const bf16_t* vp = Vbase + (size_t)l16 * T_LEN + kb + lh * 16;
#pragma unroll
        for (int dt = 0; dt < 4; ++dt) {
            v16bf av = load_a_frag(vp + (size_t)(dt * 16) * T_LEN);
            o[dt] = wmma_bf16(av, pb, o[dt]);
        }
    }

    float inv = 1.f / fmaxf(s_run, 1e-30f);
    bf16_t* orow = Ob + (size_t)(bi * T_LEN + qi) * D_MODEL + h * DK;
#pragma unroll
    for (int dt = 0; dt < 4; ++dt)
#pragma unroll
        for (int j = 0; j < 8; ++j)
            orow[dt * 16 + j + 8 * lh] = (bf16_t)(o[dt][j] * inv);
}

// ---------- workspace layout (bytes) ----------
#define O_XBF  ((size_t)0)           // 8 MB  x bf16
#define O_WQ   ((size_t)8388608)     // 2 MB
#define O_WK   ((size_t)10485760)    // 0.5 MB
#define O_WV   ((size_t)11010048)    // 0.5 MB
#define O_WO   ((size_t)11534336)    // 2 MB
#define O_QF   ((size_t)13631488)    // 16 MB  Q fp32
#define O_KF   ((size_t)30408704)    // 4 MB   K fp32
#define O_VF   ((size_t)34603008)    // 4 MB   V fp32
#define O_QBF  ((size_t)38797312)    // 8 MB   Q bf16 (post-RoPE)
#define O_KBF  ((size_t)47185920)    // 2 MB   K bf16 (post-RoPE)
#define O_VT   ((size_t)49283072)    // 2 MB   V^T bf16
#define O_ABF  ((size_t)51380224)    // 8 MB   attention out bf16

extern "C" void kernel_launch(void* const* d_in, const int* in_sizes, int n_in,
                              void* d_out, int out_size, void* d_ws, size_t ws_size,
                              hipStream_t stream) {
    const float* x   = (const float*)d_in[0];
    const float* WQ  = (const float*)d_in[1];
    const float* WK  = (const float*)d_in[2];
    const float* WV  = (const float*)d_in[3];
    const float* WO  = (const float*)d_in[4];
    const int*   pos = (const int*)d_in[5];
    float* out = (float*)d_out;

    char* ws = (char*)d_ws;
    bf16_t* xbf  = (bf16_t*)(ws + O_XBF);
    bf16_t* wqb  = (bf16_t*)(ws + O_WQ);
    bf16_t* wkb  = (bf16_t*)(ws + O_WK);
    bf16_t* wvb  = (bf16_t*)(ws + O_WV);
    bf16_t* wob  = (bf16_t*)(ws + O_WO);
    float*  qf   = (float*)(ws + O_QF);
    float*  kf   = (float*)(ws + O_KF);
    float*  vf   = (float*)(ws + O_VF);
    bf16_t* qbf  = (bf16_t*)(ws + O_QBF);
    bf16_t* kbf  = (bf16_t*)(ws + O_KBF);
    bf16_t* vt   = (bf16_t*)(ws + O_VT);
    bf16_t* abf  = (bf16_t*)(ws + O_ABF);

    const int B = 256;
    // 1) fp32 -> bf16 converts
    cvt_f32_to_bf16<<<(ROWS * D_MODEL + B - 1) / B, B, 0, stream>>>(x,  xbf, ROWS * D_MODEL);
    cvt_f32_to_bf16<<<(D_MODEL * D_MODEL + B - 1) / B, B, 0, stream>>>(WQ, wqb, D_MODEL * D_MODEL);
    cvt_f32_to_bf16<<<(NUM_GROUPS * DK * D_MODEL + B - 1) / B, B, 0, stream>>>(WK, wkb, NUM_GROUPS * DK * D_MODEL);
    cvt_f32_to_bf16<<<(NUM_GROUPS * DK * D_MODEL + B - 1) / B, B, 0, stream>>>(WV, wvb, NUM_GROUPS * DK * D_MODEL);
    cvt_f32_to_bf16<<<(D_MODEL * D_MODEL + B - 1) / B, B, 0, stream>>>(WO, wob, D_MODEL * D_MODEL);

    // 2) projections (WMMA bf16, fp32 accumulate)
    // strips = (M/16)*(N/64); 8 waves per block
    gemm_bf16_f32<<<(ROWS / 16) * (D_MODEL / 64) / 8, B, 0, stream>>>(xbf, wqb, qf, ROWS, D_MODEL, D_MODEL);
    gemm_bf16_f32<<<(ROWS / 16) * ((NUM_GROUPS * DK) / 64) / 8, B, 0, stream>>>(xbf, wkb, kf, ROWS, NUM_GROUPS * DK, D_MODEL);
    gemm_bf16_f32<<<(ROWS / 16) * ((NUM_GROUPS * DK) / 64) / 8, B, 0, stream>>>(xbf, wvb, vf, ROWS, NUM_GROUPS * DK, D_MODEL);

    // 3) RoPE + bf16, V transpose
    rope_to_bf16<<<(ROWS * (D_MODEL / 2) + B - 1) / B, B, 0, stream>>>(qf, qbf, pos, D_MODEL, ROWS * (D_MODEL / 2));
    rope_to_bf16<<<(ROWS * (NUM_GROUPS * DK / 2) + B - 1) / B, B, 0, stream>>>(kf, kbf, pos, NUM_GROUPS * DK, ROWS * (NUM_GROUPS * DK / 2));
    v_transpose_bf16<<<(ROWS * NUM_GROUPS * DK + B - 1) / B, B, 0, stream>>>(vf, vt, ROWS * NUM_GROUPS * DK);

    // 4) sliding-window flash attention (WMMA)
    dim3 agrid(T_LEN / 16 / 8, NUM_HEADS, BATCH);
    attn_kernel<<<agrid, B, 0, stream>>>(qbf, kbf, vt, abf);

    // 5) output projection -> fp32 d_out
    gemm_bf16_f32<<<(ROWS / 16) * (D_MODEL / 64) / 8, B, 0, stream>>>(abf, wob, out, ROWS, D_MODEL, D_MODEL);
}
